// Model_36550171689345
// MI455X (gfx1250) — matrix operand order repaired and checked by reading, not compiled, not run on hardware
//
#include <hip/hip_runtime.h>
#include <hip/hip_bf16.h>
#include <stdint.h>

// ---------------- model dims ----------------
static constexpr int LL  = 2;
static constexpr int DD  = 2048;
static constexpr int HN  = 16;
static constexpr int HDI = 128;
static constexpr int FFD = 5632;
static constexpr int VV  = 32000;
static constexpr int SS  = 1024;

typedef __bf16 bf16;
typedef __attribute__((ext_vector_type(16))) __bf16 v16bf;
typedef __attribute__((ext_vector_type(8)))  __bf16 v8bf;
typedef __attribute__((ext_vector_type(8)))  float  v8f;
typedef int v4i_ __attribute__((vector_size(16)));

// gfx1250 async memory->LDS path (probe via __has_builtin; VGPR-staging fallback)
#if defined(__HIP_DEVICE_COMPILE__) && \
    __has_builtin(__builtin_amdgcn_global_load_async_to_lds_b128) && \
    __has_builtin(__builtin_amdgcn_s_wait_asynccnt)
#define USE_ASYNC 1
#else
#define USE_ASYNC 0
#endif

__device__ __constant__ float NF4_LUT[16] = {
    -1.0f, -0.6961928009986877f, -0.5250730514526367f, -0.39491748809814453f,
    -0.28444138169288635f, -0.18477343022823334f, -0.09105003625154495f, 0.0f,
    0.07958029955625534f, 0.16093020141124725f, 0.24611230194568634f,
    0.33791524171829224f, 0.44070982933044434f, 0.5626170039176941f,
    0.7229568362236023f, 1.0f};

// ---------------- helper kernels ----------------
__global__ void k_embed(const int* __restrict__ ids, const float* __restrict__ embed,
                        float* __restrict__ h) {
    int s = blockIdx.x;
    long tok = ids[s];
    const float* src = embed + tok * (long)DD;
    float* dst = h + (long)s * DD;
    for (int c = threadIdx.x; c < DD; c += blockDim.x) dst[c] = src[c];
}

__global__ void k_rmsnorm(const float* __restrict__ x, const float* __restrict__ w,
                          bf16* __restrict__ out) {
    __shared__ float red[256];
    int s = blockIdx.x;
    const float* xr = x + (long)s * DD;
    float acc = 0.f;
    for (int c = threadIdx.x; c < DD; c += 256) { float v = xr[c]; acc += v * v; }
    red[threadIdx.x] = acc; __syncthreads();
    for (int o = 128; o > 0; o >>= 1) {
        if (threadIdx.x < o) red[threadIdx.x] += red[threadIdx.x + o];
        __syncthreads();
    }
    float r = rsqrtf(red[0] / (float)DD + 1e-5f);
    for (int c = threadIdx.x; c < DD; c += 256)
        out[(long)s * DD + c] = (bf16)(xr[c] * r * w[c]);
}

// vectorized NF4 dequant: 8 codes -> 8 bf16 per thread (uint2 in, uint4 out)
__global__ void k_dequant(const uint8_t* __restrict__ codes, const uint8_t* __restrict__ absq,
                          const float* __restrict__ scale, const float* __restrict__ offp,
                          bf16* __restrict__ w, long n) {
    float off = offp[0];
    long nv = n >> 3;
    for (long v = (long)blockIdx.x * 256 + threadIdx.x; v < nv; v += (long)gridDim.x * 256) {
        long e = v << 3;
        long b = e >> 6;          // BLK = 64 (same block for all 8 elems)
        long g = b >> 8;          // GRP = 256
        float am = (float)absq[b] * scale[g] + off;
        unsigned cw0 = *(const unsigned*)(codes + e);
        unsigned cw1 = *(const unsigned*)(codes + e + 4);
        union { bf16 h[8]; uint4 u; } o;
#pragma unroll
        for (int j = 0; j < 4; ++j) o.h[j]     = (bf16)(NF4_LUT[(cw0 >> (8 * j)) & 15] * am);
#pragma unroll
        for (int j = 0; j < 4; ++j) o.h[4 + j] = (bf16)(NF4_LUT[(cw1 >> (8 * j)) & 15] * am);
        *(uint4*)(w + e) = o.u;
    }
}

__global__ void k_rope_pack(const float* __restrict__ qf, const float* __restrict__ kf,
                            bf16* __restrict__ qb, bf16* __restrict__ kb) {
    int s = blockIdx.x;
    for (int c = threadIdx.x; c < DD; c += 256) {
        int hh = c >> 7, d = c & 127;
        int dr = d & 63;
        float inv = __expf(-((float)(2 * dr) / (float)HDI) * 9.210340371976184f); // ln 1e4
        float t = (float)s * inv;
        float cs = __cosf(t), sn = __sinf(t);
        long base = (long)s * DD;
        float q0 = qf[base + c], k0 = kf[base + c];
        float qo, ko;
        if (d < 64) {
            float q2 = qf[base + c + 64], k2 = kf[base + c + 64];
            qo = q0 * cs - q2 * sn; ko = k0 * cs - k2 * sn;
        } else {
            float q1 = qf[base + c - 64], k1 = kf[base + c - 64];
            qo = q0 * cs + q1 * sn; ko = k0 * cs + k1 * sn;
        }
        long idx = ((long)hh * SS + s) * HDI + d;   // [H,S,HD]
        qb[idx] = (bf16)qo; kb[idx] = (bf16)ko;
    }
}

__global__ void k_vtrans(const float* __restrict__ vf, bf16* __restrict__ vt) {
    long n = (long)SS * DD;
    for (long e = (long)blockIdx.x * 256 + threadIdx.x; e < n; e += (long)gridDim.x * 256) {
        long s = e / DD; int c = (int)(e % DD);
        int hh = c >> 7, d = c & 127;
        vt[((long)hh * HDI + d) * SS + s] = (bf16)vf[e];   // [H,HD,S]
    }
}

__global__ void k_softmax(const float* __restrict__ sc, bf16* __restrict__ pr) {
    __shared__ float red[256];
    int i = blockIdx.x;
    const float* row = sc + (long)i * SS;
    float m = -1e30f;
    for (int j = threadIdx.x; j < SS; j += 256) m = fmaxf(m, row[j]);
    red[threadIdx.x] = m; __syncthreads();
    for (int o = 128; o > 0; o >>= 1) {
        if (threadIdx.x < o) red[threadIdx.x] = fmaxf(red[threadIdx.x], red[threadIdx.x + o]);
        __syncthreads();
    }
    m = red[0]; __syncthreads();
    float ssum = 0.f;
    for (int j = threadIdx.x; j < SS; j += 256) ssum += __expf(row[j] - m);
    red[threadIdx.x] = ssum; __syncthreads();
    for (int o = 128; o > 0; o >>= 1) {
        if (threadIdx.x < o) red[threadIdx.x] += red[threadIdx.x + o];
        __syncthreads();
    }
    float inv = 1.0f / red[0];
    for (int j = threadIdx.x; j < SS; j += 256)
        pr[(long)i * SS + j] = (bf16)(__expf(row[j] - m) * inv);
}

__global__ void k_silumul(const float* __restrict__ g, const float* __restrict__ u,
                          bf16* __restrict__ out, long n) {
    for (long e = (long)blockIdx.x * 256 + threadIdx.x; e < n; e += (long)gridDim.x * 256) {
        float x = g[e];
        float sl = x / (1.0f + __expf(-x));
        out[e] = (bf16)(sl * u[e]);
    }
}

__global__ void k_cvt(const float* __restrict__ x, bf16* __restrict__ y, long n) {
    for (long e = (long)blockIdx.x * 256 + threadIdx.x; e < n; e += (long)gridDim.x * 256)
        y[e] = (bf16)x[e];
}

// ------- fragment helpers -------
__device__ __forceinline__ v16bf cat16(v8bf a, v8bf b) {
    return __builtin_shufflevector(a, b, 0, 1, 2, 3, 4, 5, 6, 7,
                                         8, 9, 10, 11, 12, 13, 14, 15);
}

#if USE_ASYNC
__device__ __forceinline__ void async_ld16(const bf16* g, bf16* l) {
    // signature (from hipcc diagnostic): param0 = int4-vector pointer in global AS
    __builtin_amdgcn_global_load_async_to_lds_b128(
        (__attribute__((address_space(1))) v4i_*)g,
        (__attribute__((address_space(3))) v4i_*)l, 0, 0);
}
#endif

// ---------------- WMMA GEMM: C[M,N] = A[M,K](bf16) * B[N,K]^T ----------------
// grid = (N/64, M/128); block = 256 (8 waves). Wave owns 32x32 of C (4 accums).
// Double-buffered LDS; async memory->LDS DMA when available.
// ADD: C += acc. SCORES: acc*alpha + causal/attn-mask bias.
// BF32: B source fp32 (embed), converted to bf16 during staging (VGPR path).
template <int ADD, int SCORES, int BF32>
__global__ __launch_bounds__(256) void k_gemm(
    const bf16* __restrict__ A, const void* __restrict__ Bvoid, float* __restrict__ C,
    int N, int K, int lda, int ldb, int ldc,
    float alpha, const int* __restrict__ mask) {
    __shared__ bf16 As[2][128][72];   // 36.9 KB (144B row stride, 16B aligned)
    __shared__ bf16 Bs[2][64][72];    // 18.4 KB

    const int tid = threadIdx.x;
    const int bm = blockIdx.y, bn = blockIdx.x;
    const int wv = tid >> 5, lane = tid & 31;
    const int m0 = (wv & 3) * 32;     // wave row group
    const int n0 = (wv >> 2) * 32;    // wave col group
    const int lr = lane & 15, lhi = lane >> 4;

    v8f acc[2][2] = {{{}, {}}, {{}, {}}};

    const bf16*  Bb = (const bf16*)Bvoid;
    const float* Bf = (const float*)Bvoid;

    const int srow = tid >> 3;        // 0..31
    const int sch  = tid & 7;         // 16B chunk in 64-col row

    // stage one 128x64 A tile + 64x64 B tile into buffer sb
    auto stage = [&](int sb, int kt) {
#if USE_ASYNC
#pragma unroll
        for (int j = 0; j < 4; ++j) {
            int r = srow + 32 * j;
            async_ld16(A + ((long)bm * 128 + r) * lda + kt + 8 * sch, &As[sb][r][8 * sch]);
        }
        if (!BF32) {
#pragma unroll
            for (int j = 0; j < 2; ++j) {
                int r = srow + 32 * j;
                async_ld16(Bb + ((long)bn * 64 + r) * ldb + kt + 8 * sch, &Bs[sb][r][8 * sch]);
            }
        }
#else
        uint4 ra[4];
#pragma unroll
        for (int j = 0; j < 4; ++j)
            ra[j] = *(const uint4*)(A + ((long)bm * 128 + srow + 32 * j) * lda + kt + 8 * sch);
#pragma unroll
        for (int j = 0; j < 4; ++j)
            *(uint4*)&As[sb][srow + 32 * j][8 * sch] = ra[j];
        if (!BF32) {
            uint4 rb[2];
#pragma unroll
            for (int j = 0; j < 2; ++j)
                rb[j] = *(const uint4*)(Bb + ((long)bn * 64 + srow + 32 * j) * ldb + kt + 8 * sch);
#pragma unroll
            for (int j = 0; j < 2; ++j)
                *(uint4*)&Bs[sb][srow + 32 * j][8 * sch] = rb[j];
        }
#endif
        if (BF32) {   // fp32 B: load + convert through VGPRs
#pragma unroll
            for (int j = 0; j < 2; ++j) {
                int r = srow + 32 * j;
                const float* src = Bf + ((long)bn * 64 + r) * ldb + kt + 8 * sch;
                float4 f0 = *(const float4*)src;
                float4 f1 = *(const float4*)(src + 4);
                bf16* d = &Bs[sb][r][8 * sch];
                d[0] = (bf16)f0.x; d[1] = (bf16)f0.y; d[2] = (bf16)f0.z; d[3] = (bf16)f0.w;
                d[4] = (bf16)f1.x; d[5] = (bf16)f1.y; d[6] = (bf16)f1.z; d[7] = (bf16)f1.w;
            }
        }
    };

    int buf = 0;
    stage(0, 0);
    for (int kt = 0; kt < K; kt += 64) {
        const int nbuf = buf ^ 1;
        const bool has_next = (kt + 64) < K;
        if (has_next) stage(nbuf, kt + 64);
#if USE_ASYNC
        if (has_next) {                 // wait for current tile only (in-order done)
            if (BF32) __builtin_amdgcn_s_wait_asynccnt(4);
            else      __builtin_amdgcn_s_wait_asynccnt(6);
        } else {
            __builtin_amdgcn_s_wait_asynccnt(0);
        }
#endif
        __syncthreads();                // current tile visible to all waves

#pragma unroll
        for (int kc = 0; kc < 64; kc += 32) {
            v16bf af[2], bfr[2];
#pragma unroll
            for (int mi = 0; mi < 2; ++mi) {
                const bf16* ap = &As[buf][m0 + mi * 16 + lr][kc + lhi * 8];
                af[mi] = cat16(*(const v8bf*)ap, *(const v8bf*)(ap + 16));
            }
#pragma unroll
            for (int ni = 0; ni < 2; ++ni) {
                const bf16* bp = &Bs[buf][n0 + ni * 16 + lr][kc + lhi * 8];
                bfr[ni] = cat16(*(const v8bf*)bp, *(const v8bf*)(bp + 16));
            }
#pragma unroll
            for (int mi = 0; mi < 2; ++mi)
#pragma unroll
                for (int ni = 0; ni < 2; ++ni)
                    acc[mi][ni] = __builtin_amdgcn_wmma_f32_16x16x32_bf16(
                        false, af[mi], false, bfr[ni], (short)0, acc[mi][ni], false, false);
        }
        __syncthreads();                // done reading buf; next iter may overwrite it
        buf = nbuf;
    }

#pragma unroll
    for (int mi = 0; mi < 2; ++mi)
#pragma unroll
        for (int ni = 0; ni < 2; ++ni)
#pragma unroll
            for (int i = 0; i < 8; ++i) {
                int gm = bm * 128 + m0 + mi * 16 + i + lhi * 8;
                int gn = bn * 64 + n0 + ni * 16 + lr;
                float val = acc[mi][ni][i];
                if (SCORES) {
                    bool ok = (gn <= gm) && (mask[gn] > 0);
                    val = val * alpha + (ok ? 0.0f : -1e9f);
                }
                long idx = (long)gm * ldc + gn;
                if (ADD) C[idx] += val; else C[idx] = val;
            }
}

// ---------------- orchestration ----------------
extern "C" void kernel_launch(void* const* d_in, const int* in_sizes, int n_in,
                              void* d_out, int out_size, void* d_ws, size_t ws_size,
                              hipStream_t stream) {
    (void)in_sizes; (void)n_in; (void)out_size; (void)ws_size;

    const int* input_ids = (const int*)d_in[0];
    const int* attn_mask = (const int*)d_in[1];
    const uint8_t* codes[7]; const uint8_t* absq[7];
    for (int p = 0; p < 7; ++p) {
        codes[p] = (const uint8_t*)d_in[2 + 2 * p];
        absq[p]  = (const uint8_t*)d_in[3 + 2 * p];
    }
    const float* embed = (const float*)d_in[16];
    const float* ln1 = (const float*)d_in[17];
    const float* ln2 = (const float*)d_in[18];
    const float* lnf = (const float*)d_in[19];
    const float* qsc[7]; const float* qoff[7];
    for (int p = 0; p < 7; ++p) {
        qsc[p]  = (const float*)d_in[20 + 2 * p];
        qoff[p] = (const float*)d_in[21 + 2 * p];
    }
    // proj order: 0=q 1=k 2=v 3=o 4=gate 5=up 6=down
    const long OUTD[7] = {DD, DD, DD, DD, FFD, FFD, DD};
    const long IND[7]  = {DD, DD, DD, DD, DD,  DD,  FFD};

    char* ws = (char*)d_ws;
    size_t off = 0;
    auto carve = [&](size_t bytes) { char* r = ws + off; off = (off + bytes + 255) & ~(size_t)255; return r; };
    bf16*  wdeq = (bf16*) carve((size_t)FFD * DD * 2);
    bf16*  xb   = (bf16*) carve((size_t)SS * DD * 2);
    float* h    = (float*)carve((size_t)SS * DD * 4);
    float* t0   = (float*)carve((size_t)SS * FFD * 4);
    float* t1   = (float*)carve((size_t)SS * FFD * 4);
    bf16*  qb   = (bf16*) carve((size_t)SS * DD * 2);
    bf16*  kb   = (bf16*) carve((size_t)SS * DD * 2);
    bf16*  vt   = (bf16*) carve((size_t)SS * DD * 2);
    float* sc   = (float*)carve((size_t)SS * SS * 4);
    bf16*  pr   = (bf16*) carve((size_t)SS * SS * 2);
    bf16*  gs   = (bf16*) carve((size_t)SS * FFD * 2);
    bf16*  ctxb = (bf16*) carve((size_t)SS * DD * 2);

    auto dequant = [&](int p, int l) {
        long n = OUTD[p] * IND[p];
        long nb = n >> 6, ng = nb >> 8;
        k_dequant<<<dim3((unsigned)(n >> 11)), 256, 0, stream>>>(
            codes[p] + (size_t)l * n, absq[p] + (size_t)l * nb,
            qsc[p] + (size_t)l * ng, qoff[p] + l, wdeq, n);
    };

    const float scale = 0.08838834764831845f;  // 1/sqrt(128)

    k_embed<<<SS, 256, 0, stream>>>(input_ids, embed, h);

    for (int l = 0; l < LL; ++l) {
        k_rmsnorm<<<SS, 256, 0, stream>>>(h, ln1 + (size_t)l * DD, xb);

        dequant(0, l);   // q
        k_gemm<0, 0, 0><<<dim3(DD / 64, SS / 128), 256, 0, stream>>>(
            xb, wdeq, t0, DD, DD, DD, DD, DD, 1.0f, nullptr);
        dequant(1, l);   // k
        k_gemm<0, 0, 0><<<dim3(DD / 64, SS / 128), 256, 0, stream>>>(
            xb, wdeq, t1, DD, DD, DD, DD, DD, 1.0f, nullptr);
        k_rope_pack<<<SS, 256, 0, stream>>>(t0, t1, qb, kb);

        dequant(2, l);   // v
        k_gemm<0, 0, 0><<<dim3(DD / 64, SS / 128), 256, 0, stream>>>(
            xb, wdeq, t0, DD, DD, DD, DD, DD, 1.0f, nullptr);
        k_vtrans<<<8192, 256, 0, stream>>>(t0, vt);

        for (int hh = 0; hh < HN; ++hh) {
            // scores = scale * Q Kt + bias   (M=N=S, K=HD)
            k_gemm<0, 1, 0><<<dim3(SS / 64, SS / 128), 256, 0, stream>>>(
                qb + (size_t)hh * SS * HDI, kb + (size_t)hh * SS * HDI, sc,
                SS, HDI, HDI, HDI, SS, scale, attn_mask);
            k_softmax<<<SS, 256, 0, stream>>>(sc, pr);
            // ctx[:, hh*HD : hh*HD+HD] = P @ Vt^T   (M=S, N=HD, K=S)
            k_gemm<0, 0, 0><<<dim3(HDI / 64, SS / 128), 256, 0, stream>>>(
                pr, vt + (size_t)hh * HDI * SS, t0 + (size_t)hh * HDI,
                HDI, SS, SS, SS, DD, 1.0f, nullptr);
        }
        k_cvt<<<8192, 256, 0, stream>>>(t0, ctxb, (long)SS * DD);

        dequant(3, l);   // o (residual add)
        k_gemm<1, 0, 0><<<dim3(DD / 64, SS / 128), 256, 0, stream>>>(
            ctxb, wdeq, h, DD, DD, DD, DD, DD, 1.0f, nullptr);

        k_rmsnorm<<<SS, 256, 0, stream>>>(h, ln2 + (size_t)l * DD, xb);

        dequant(4, l);   // gate
        k_gemm<0, 0, 0><<<dim3(FFD / 64, SS / 128), 256, 0, stream>>>(
            xb, wdeq, t0, FFD, DD, DD, DD, FFD, 1.0f, nullptr);
        dequant(5, l);   // up
        k_gemm<0, 0, 0><<<dim3(FFD / 64, SS / 128), 256, 0, stream>>>(
            xb, wdeq, t1, FFD, DD, DD, DD, FFD, 1.0f, nullptr);
        k_silumul<<<22528, 256, 0, stream>>>(t0, t1, gs, (long)SS * FFD);

        dequant(6, l);   // down (residual add)
        k_gemm<1, 0, 0><<<dim3(DD / 64, SS / 128), 256, 0, stream>>>(
            gs, wdeq, h, DD, FFD, FFD, FFD, DD, 1.0f, nullptr);
    }

    k_rmsnorm<<<SS, 256, 0, stream>>>(h, lnf, xb);
    // lm_head: out = xb @ embed^T  (B staged from fp32)
    k_gemm<0, 0, 1><<<dim3(VV / 64, SS / 128), 256, 0, stream>>>(
        xb, embed, (float*)d_out, VV, DD, DD, DD, VV, 1.0f, nullptr);
}
